// Refine_88871463289316
// MI455X (gfx1250) — matrix operand order repaired and checked by reading, not compiled, not run on hardware
//
#include <hip/hip_runtime.h>
#include <hip/hip_bf16.h>
#include <math.h>

// ---------------------------------------------------------------------------
// MI455X / gfx1250, wave32. bf16 WMMA (16x16x32, f32 accum) for both GEMMs.
// All matrix operands staged in fragment-linear layout so each fragment is a
// single v16bf (2 x b128) load per lane.
// ---------------------------------------------------------------------------

typedef __attribute__((ext_vector_type(16))) __bf16 v16bf;
typedef __attribute__((ext_vector_type(8)))  float  v8f;

#define B_   16
#define C_   256
#define H_   128
#define W_   128
#define P_   (H_ * W_)      // 16384 pixels per image
#define N_   80             // classes (5 tiles of 16)
#define TP   64             // pixels per block tile (4 waves x 16 rows)
#define FTS  65             // LDS pixel stride (bank-conflict pad)
#define EPSN 1e-12f

#define CT_ELEMS (5 * 8 * 32 * 16)        // 20480 bf16, frag-linear sim B
#define WT_ELEMS (4 * 4 * 8 * 32 * 16)    // 65536 bf16, frag-linear conv B

__device__ __forceinline__ unsigned bfbits(float f) {
  unsigned u = __float_as_uint(f);
  return (u + 0x7fffu + ((u >> 16) & 1u)) >> 16;  // RNE, low 16 bits valid
}

// ---------------------------------------------------------------------------
// Zero the scatter buffers (sum_x and count) each call.
// ---------------------------------------------------------------------------
__global__ void zero_kernel(float* __restrict__ sumx, float* __restrict__ cnt) {
  int i = blockIdx.x * 256 + threadIdx.x;
  if (i < B_ * N_ * C_) sumx[i] = 0.f;
  if (i < B_ * N_)      cnt[i]  = 0.f;
}

// ---------------------------------------------------------------------------
// prep_norm: 1/max(||centroid||,eps) per class.
// ---------------------------------------------------------------------------
__global__ void prep_norm(const float* __restrict__ cen, float* __restrict__ crinv) {
  __shared__ float red[256];
  const int t = threadIdx.x, n = blockIdx.x;
  float v = cen[n * C_ + t];
  red[t] = v * v;
  __syncthreads();
  #pragma unroll
  for (int s = 128; s > 0; s >>= 1) {
    if (t < s) red[t] += red[t + s];
    __syncthreads();
  }
  if (t == 0) crinv[n] = 1.0f / fmaxf(sqrtf(red[0]), EPSN);
}

// ---------------------------------------------------------------------------
// prep_pack: write normalized centroids and conv weight into frag-linear
// bf16 B-matrix order: flat = ((tile*8 + kk)*32 + lane)*16 + j,
// with K = kk*32 + j + 16*(lane>>4), col = tile*16 + (lane&15).
// ---------------------------------------------------------------------------
__global__ void prep_pack(const float* __restrict__ cen,
                          const float* __restrict__ crinv,
                          const float* __restrict__ fcw,
                          unsigned short* __restrict__ cTf,
                          unsigned short* __restrict__ wTf) {
  int i = blockIdx.x * 256 + threadIdx.x;
  if (i < CT_ELEMS) {
    int j = i & 15, lane = (i >> 4) & 31, kk = (i >> 9) & 7, nt = i >> 12;
    int K = kk * 32 + j + 16 * (lane >> 4);
    int n = nt * 16 + (lane & 15);
    cTf[i] = (unsigned short)bfbits(cen[n * C_ + K] * crinv[n]);
  } else {
    int ii = i - CT_ELEMS;
    if (ii < WT_ELEMS) {
      int j = ii & 15, lane = (ii >> 4) & 31, kk = (ii >> 9) & 7;
      int nt = (ii >> 12) & 3, og = ii >> 14;
      int K = kk * 32 + j + 16 * (lane >> 4);
      int o = og * 64 + nt * 16 + (lane & 15);
      wTf[ii] = (unsigned short)bfbits(fcw[o * C_ + K]);  // fc_w is [O][C]
    }
  }
}

// ---------------------------------------------------------------------------
// Assign: per 64-pixel tile: load features, 1/||f||, stage normalized bf16
// A-frags in LDS, sim GEMM (WMMA), argmax over 80 classes, scatter sums.
// ---------------------------------------------------------------------------
__global__ void __launch_bounds__(128) assign_kernel(
    const float* __restrict__ feat, const unsigned short* __restrict__ cTf,
    int* __restrict__ idxg, float* __restrict__ sumx, float* __restrict__ cnt) {
  __shared__ float ft[C_ * FTS];
  __shared__ alignas(32) unsigned int abuf[4 * 8 * 32 * 8];  // frag-linear bf16 A
  __shared__ float rinv[TP];
  __shared__ float part[128];
  __shared__ int   cls[TP];

  const int t  = threadIdx.x;
  const int b  = blockIdx.x >> 8;
  const int p0 = (blockIdx.x & 255) * TP;
  const float* fb = feat + (size_t)b * C_ * P_ + p0;

  // --- load 64px x 256ch tile (channel pairs), fused sum-of-squares -------
  const int p  = t & (TP - 1);
  const int ch = t >> 6;               // half of channel range per thread
  float ss = 0.f;
  #pragma unroll 4
  for (int k = 0; k < 64; ++k) {
    int c0 = ch * 128 + 2 * k;
    float v0 = fb[(size_t)c0 * P_ + p];
    float v1 = fb[(size_t)(c0 + 1) * P_ + p];
    ft[c0 * FTS + p] = v0;
    ft[(c0 + 1) * FTS + p] = v1;
    ss += v0 * v0 + v1 * v1;
  }
  part[t] = ss;
  __syncthreads();
  if (t < TP) rinv[t] = 1.0f / fmaxf(sqrtf(part[t] + part[t + TP]), EPSN);
  __syncthreads();

  // --- stage normalized bf16 A-fragments (packed channel pairs) -----------
  {
    const float rn = rinv[p];
    const int w = p >> 4, m = p & 15;
    #pragma unroll 4
    for (int k = 0; k < 64; ++k) {
      int c0 = ch * 128 + 2 * k;
      int kk = c0 >> 5, Kp = c0 & 31;
      int hf = Kp >> 4;
      int r  = (Kp & 15) >> 1;
      int lane = hf * 16 + m;
      unsigned lo = bfbits(ft[c0 * FTS + p] * rn);
      unsigned hi = bfbits(ft[(c0 + 1) * FTS + p] * rn);
      abuf[((w * 8 + kk) * 32 + lane) * 8 + r] = lo | (hi << 16);
    }
  }
  __syncthreads();

  // --- sim GEMM: D[16px x 80cls] per wave ---------------------------------
  const int wv   = t >> 5;
  const int lane = t & 31;
  const int nl   = lane & 15;
  const int hf   = lane >> 4;
  const v16bf* Af = (const v16bf*)abuf;
  const v16bf* Bc = (const v16bf*)cTf;

  v8f acc[5];
  #pragma unroll
  for (int nt = 0; nt < 5; ++nt)
    #pragma unroll
    for (int v = 0; v < 8; ++v) acc[nt][v] = 0.f;

  #pragma unroll
  for (int kk = 0; kk < 8; ++kk) {
    v16bf a = Af[(wv * 8 + kk) * 32 + lane];
    #pragma unroll
    for (int nt = 0; nt < 5; ++nt) {
      v16bf bb = Bc[(nt * 8 + kk) * 32 + lane];
      acc[nt] = __builtin_amdgcn_wmma_f32_16x16x32_bf16(
          false, a, false, bb, (short)0, acc[nt], false, false);
    }
  }

  // --- argmax over classes (C/D layout: VGPR v = row v+8*hf, lane = col) --
  #pragma unroll
  for (int v = 0; v < 8; ++v) {
    float bv = acc[0][v];
    int   bi = nl;
    #pragma unroll
    for (int nt = 1; nt < 5; ++nt) {
      float cv = acc[nt][v];
      if (cv > bv) { bv = cv; bi = nt * 16 + nl; }
    }
    #pragma unroll
    for (int d = 1; d < 16; d <<= 1) {  // reduce within each 16-lane group
      float ov = __shfl_xor(bv, d, 32);
      int   oi = __shfl_xor(bi, d, 32);
      if (ov > bv || (ov == bv && oi < bi)) { bv = ov; bi = oi; }
    }
    if (nl == 0) {
      int pp = wv * 16 + v + 8 * hf;
      cls[pp] = bi;
      idxg[(size_t)b * P_ + p0 + pp] = bi;
    }
  }
  __syncthreads();

  // --- scatter per-(image,class) raw feature sums + counts ----------------
  const int n_p = cls[p];
  float* dst = sumx + (size_t)(b * N_ + n_p) * C_;
  #pragma unroll 4
  for (int k = 0; k < 64; ++k) {
    int c0 = ch * 128 + 2 * k;
    atomicAdd(dst + c0,     ft[c0 * FTS + p]);
    atomicAdd(dst + c0 + 1, ft[(c0 + 1) * FTS + p]);
  }
  if (t < TP) atomicAdd(cnt + b * N_ + cls[t], 1.0f);
}

// ---------------------------------------------------------------------------
// cent = sum_x / max(count, 1)
// ---------------------------------------------------------------------------
__global__ void cent_kernel(const float* __restrict__ sumx,
                            const float* __restrict__ cnt,
                            float* __restrict__ cent) {
  int i = blockIdx.x * 256 + threadIdx.x;
  cent[i] = sumx[i] / fmaxf(cnt[i >> 8], 1.0f);
}

// ---------------------------------------------------------------------------
// Refine: gather class mean, alpha blend straight into bf16 A-frag staging,
// fused 1x1-conv GEMM + bias + ReLU.
// ---------------------------------------------------------------------------
__global__ void __launch_bounds__(128) refine_kernel(
    const float* __restrict__ feat, const int* __restrict__ idxg,
    const float* __restrict__ cent, const unsigned short* __restrict__ wTf,
    const float* __restrict__ fcb, float* __restrict__ out) {
  __shared__ float ft[C_ * FTS];
  __shared__ alignas(32) unsigned int abuf[4 * 8 * 32 * 8];  // frag-linear bf16 x
  __shared__ float part[128];
  __shared__ float alph[TP];

  const int t  = threadIdx.x;
  const int b  = blockIdx.x >> 8;
  const int p0 = (blockIdx.x & 255) * TP;
  const float* fb = feat + (size_t)b * C_ * P_ + p0;

  const int p  = t & (TP - 1);
  const int ch = t >> 6;
  #pragma unroll 4
  for (int k = 0; k < 64; ++k) {
    int c0 = ch * 128 + 2 * k;
    ft[c0 * FTS + p]       = fb[(size_t)c0 * P_ + p];
    ft[(c0 + 1) * FTS + p] = fb[(size_t)(c0 + 1) * P_ + p];
  }
  __syncthreads();

  // alpha = exp(-mean((cent[idx] - f)^2))
  const int n_p = idxg[(size_t)b * P_ + p0 + p];
  const float* cen = cent + (size_t)(b * N_ + n_p) * C_;
  float s = 0.f;
  #pragma unroll 4
  for (int k = 0; k < 64; ++k) {
    int c0 = ch * 128 + 2 * k;
    float d0 = cen[c0]     - ft[c0 * FTS + p];
    float d1 = cen[c0 + 1] - ft[(c0 + 1) * FTS + p];
    s += d0 * d0 + d1 * d1;
  }
  part[t] = s;
  __syncthreads();
  if (t < TP) alph[t] = __expf(-(part[t] + part[t + TP]) * (1.0f / C_));
  __syncthreads();

  // x = f + alpha*(cent - f), packed bf16 pairs straight into frag staging
  {
    const float al = alph[p];
    const int w = p >> 4, m = p & 15;
    #pragma unroll 4
    for (int k = 0; k < 64; ++k) {
      int c0 = ch * 128 + 2 * k;
      int kk = c0 >> 5, Kp = c0 & 31;
      int hf = Kp >> 4;
      int r  = (Kp & 15) >> 1;
      int lane = hf * 16 + m;
      float f0 = ft[c0 * FTS + p];
      float f1 = ft[(c0 + 1) * FTS + p];
      unsigned lo = bfbits(f0 + al * (cen[c0] - f0));
      unsigned hi = bfbits(f1 + al * (cen[c0 + 1] - f1));
      abuf[((w * 8 + kk) * 32 + lane) * 8 + r] = lo | (hi << 16);
    }
  }
  __syncthreads();

  // 1x1 conv GEMM: out[16px x 256o] per wave, bias seeded into accumulator
  const int wv = t >> 5, lane = t & 31, nl = lane & 15, hf = lane >> 4;
  const v16bf* Af = (const v16bf*)abuf;
  const v16bf* Bw = (const v16bf*)wTf;

  for (int og = 0; og < 4; ++og) {
    v8f acc[4];
    #pragma unroll
    for (int nt = 0; nt < 4; ++nt) {
      float bias = fcb[og * 64 + nt * 16 + nl];
      #pragma unroll
      for (int v = 0; v < 8; ++v) acc[nt][v] = bias;
    }
    #pragma unroll
    for (int kk = 0; kk < 8; ++kk) {
      v16bf a = Af[(wv * 8 + kk) * 32 + lane];
      #pragma unroll
      for (int nt = 0; nt < 4; ++nt) {
        v16bf bb = Bw[((og * 4 + nt) * 8 + kk) * 32 + lane];
        acc[nt] = __builtin_amdgcn_wmma_f32_16x16x32_bf16(
            false, a, false, bb, (short)0, acc[nt], false, false);
      }
    }
    #pragma unroll
    for (int nt = 0; nt < 4; ++nt) {
      int o = og * 64 + nt * 16 + nl;
      #pragma unroll
      for (int v = 0; v < 8; ++v) {
        int pp = p0 + wv * 16 + v + 8 * hf;
        out[(size_t)(b * C_ + o) * P_ + pp] = fmaxf(acc[nt][v], 0.0f);
      }
    }
  }
}

// ---------------------------------------------------------------------------
// Host launcher
// ---------------------------------------------------------------------------
extern "C" void kernel_launch(void* const* d_in, const int* in_sizes, int n_in,
                              void* d_out, int out_size, void* d_ws, size_t ws_size,
                              hipStream_t stream) {
  (void)in_sizes; (void)n_in; (void)out_size; (void)ws_size;
  const float* feat = (const float*)d_in[0];   // [16,256,128,128]
  const float* cen  = (const float*)d_in[1];   // [80,256]
  const float* fcw  = (const float*)d_in[2];   // [256,256,1,1]
  const float* fcb  = (const float*)d_in[3];   // [256]
  float* out = (float*)d_out;

  char* ws = (char*)d_ws;
  float* sumx = (float*)(ws);                       // 1,310,720 B
  float* cent = (float*)(ws + 1310720);             // 1,310,720 B
  float* cnt  = (float*)(ws + 2621440);             //     5,120 B
  int*   idxg = (int*)  (ws + 2626560);             // 1,048,576 B
  float* crinv = (float*)(ws + 3675136);            //       320 B
  unsigned short* cTf = (unsigned short*)(ws + 3675520);  //  40,960 B (32B aligned)
  unsigned short* wTf = (unsigned short*)(ws + 3716480);  // 131,072 B (32B aligned)

  zero_kernel<<<1280, 256, 0, stream>>>(sumx, cnt);
  prep_norm  <<<N_, 256, 0, stream>>>(cen, crinv);
  prep_pack  <<<(CT_ELEMS + WT_ELEMS) / 256, 256, 0, stream>>>(cen, crinv, fcw, cTf, wTf);
  assign_kernel<<<B_ * (P_ / TP), 128, 0, stream>>>(feat, cTf, idxg, sumx, cnt);
  cent_kernel  <<<1280, 256, 0, stream>>>(sumx, cnt, cent);
  refine_kernel<<<B_ * (P_ / TP), 128, 0, stream>>>(feat, idxg, cent, wTf, fcb, out);
}
